// SS3D_block_23192823398804
// MI455X (gfx1250) — matrix operand order repaired and checked by reading, not compiled, not run on hardware
//
#include <hip/hip_runtime.h>
#include <hip/hip_bf16.h>
#include <math.h>

typedef __attribute__((ext_vector_type(16))) _Float16 v16h;
typedef __attribute__((ext_vector_type(8)))  _Float16 v8h;
typedef __attribute__((ext_vector_type(4)))  _Float16 v4h;
typedef __attribute__((ext_vector_type(8)))  float    v8f;

#define KDIR   12
#define DCH    128
#define NST    16
#define LSEQ   4096
#define CHUNK  256
#define NCHUNK 16

__device__ __constant__ int d_ORD[6][3] = {
  {1,2,3},{1,3,2},{3,2,1},{3,1,2},{2,1,3},{2,3,1}
};

// Map sequence index l of direction k to spatial linear index (h*256+w*16+len).
// All spatial dims are 16; odd k flips all three axes (ORDERS permute {1,2,3}).
__device__ __forceinline__ int pos_from_l(int k, int l) {
  int o = k >> 1;
  int i = (l >> 8) & 15, j = (l >> 4) & 15, m = l & 15;
  int c[4];
  c[d_ORD[o][0]] = i; c[d_ORD[o][1]] = j; c[d_ORD[o][2]] = m;
  if (k & 1) { c[1] = 15 - c[1]; c[2] = 15 - c[2]; c[3] = 15 - c[3]; }
  return (c[1] << 8) | (c[2] << 4) | c[3];
}

__device__ __forceinline__ v16h cat8(v8h lo, v8h hi8) {
  return __builtin_shufflevector(lo, hi8, 0,1,2,3,4,5,6,7,8,9,10,11,12,13,14,15);
}

// ---------------------------------------------------------------------------
// Kernel 1: per (k, 16-row l-tile) fused projection.
//   GEMM1 (WMMA f16->f32): out[l][c] = sum_d x_seq[l][d] * W[c][d], c in [0,40)
//   split -> dts (c<8), Bs (8..24), Cs (24..40)
//   GEMM2 (WMMA): raw[l][d] = sum_r dts[l][r] * dtw[d][r];  delta = softplus(raw+bias)
// One wave per 16-row tile; 8 waves/block; 32 blocks per direction.
// All LDS operand tiles stored K-contiguous so each WMMA fragment is built
// from 2 aligned ds_load_b128 (per the documented 16-bit lane layouts).
// ---------------------------------------------------------------------------
__global__ __launch_bounds__(256)
void proj_kernel(const float* __restrict__ x,        // (4096,128)
                 const float* __restrict__ xproj,    // (12,40,128)
                 const float* __restrict__ dtw,      // (12,128,8)
                 const float* __restrict__ dtb,      // (12,128)
                 float* __restrict__ deltabuf,       // (12,4096,128)
                 float* __restrict__ Bsbuf,          // (12,4096,16)
                 float* __restrict__ Csbuf)          // (12,4096,16)
{
  __shared__ __align__(16) _Float16 sW2[48][128];    // [c][d], c padded 40->48
  __shared__ __align__(16) _Float16 sDtwT[128][8];   // [d][r]
  __shared__ float                  sBias[128];
  __shared__ __align__(16) _Float16 sXh[8][16][128]; // per-wave x tile, f16
  __shared__ __align__(16) _Float16 sDts[8][16][8];  // per-wave dts tile

  const int tid = threadIdx.x;
  const int k   = blockIdx.x >> 5;      // 32 blocks per direction
  const int blk = blockIdx.x & 31;

  // Stage x_proj_weight transposed-to-[c][d] (K=d contiguous), f16, float4 reads.
  for (int t4 = tid; t4 < 48 * 32; t4 += 256) {
    int c = t4 >> 5, d4 = (t4 & 31) * 4;
    v4h hv = {};
    if (c < 40) {
      const float4 f = *(const float4*)&xproj[((size_t)k * 40 + c) * 128 + d4];
      hv[0] = (_Float16)f.x; hv[1] = (_Float16)f.y;
      hv[2] = (_Float16)f.z; hv[3] = (_Float16)f.w;
    }
    *(v4h*)&sW2[c][d4] = hv;
  }
  // dt_projs_weight is already (d, r): keep [d][r] (K=r contiguous per d row).
  for (int t = tid; t < 128 * 8; t += 256) {
    int d = t >> 3, r = t & 7;
    sDtwT[d][r] = (_Float16)dtw[((size_t)k * 128 + d) * 8 + r];
  }
  if (tid < 128) sBias[tid] = dtb[(size_t)k * 128 + tid];
  __syncthreads();

  const int w     = tid >> 5;
  const int lane  = tid & 31;
  const int row   = lane & 15;
  const int hi    = lane >> 4;
  const int col16 = lane & 15;
  const int l0    = (blk * 8 + w) * 16;

  // Gather this wave's 16x128 x-tile (float4 loads, v4h LDS stores).
  {
    int pos = pos_from_l(k, l0 + row);
    const float* xr = x + (size_t)pos * 128;
    int dbase = hi * 64;
    #pragma unroll
    for (int i = 0; i < 16; ++i) {
      const float4 f = *(const float4*)&xr[dbase + 4 * i];
      v4h hv;
      hv[0] = (_Float16)f.x; hv[1] = (_Float16)f.y;
      hv[2] = (_Float16)f.z; hv[3] = (_Float16)f.w;
      *(v4h*)&sXh[w][row][dbase + 4 * i] = hv;
    }
  }
  __syncthreads();

  // A fragments (16x32 f16): halves 0-7 = K hi*8..+7, halves 8-15 = K 16+hi*8..+7
  // -> two contiguous 16B runs per chunk.
  v16h afrag[4];
  #pragma unroll
  for (int ch = 0; ch < 4; ++ch) {
    v8h lo  = *(const v8h*)&sXh[w][row][ch * 32 + hi * 8];
    v8h hi8 = *(const v8h*)&sXh[w][row][ch * 32 + 16 + hi * 8];
    afrag[ch] = cat8(lo, hi8);
  }

  // GEMM1: 3 column tiles (c 0-15, 16-31, 32-47), K = 128 in 4 chunks of 32.
  #pragma unroll
  for (int ct = 0; ct < 3; ++ct) {
    v8f acc = {};
    #pragma unroll
    for (int ch = 0; ch < 4; ++ch) {
      // B frag (32x16): halves 0-7 = K hi*8..+7, halves 8-15 = K 16+hi*8..+7 (matches A).
      v8h b0 = *(const v8h*)&sW2[ct * 16 + col16][ch * 32 + hi * 8];
      v8h b1 = *(const v8h*)&sW2[ct * 16 + col16][ch * 32 + 16 + hi * 8];
      v16h bfrag = cat8(b0, b1);
      acc = __builtin_amdgcn_wmma_f32_16x16x32_f16(false, afrag[ch], false, bfrag,
                                                   (short)0, acc, false, false);
    }
    int c = ct * 16 + col16;
    #pragma unroll
    for (int r = 0; r < 8; ++r) {
      int l = l0 + r + hi * 8;                 // C/D layout: lanes>=16 hold M=8+r
      float val = acc[r];
      if (ct == 0 && col16 < 8) sDts[w][r + hi * 8][col16] = (_Float16)val;
      if (c >= 8 && c < 24)  Bsbuf[((size_t)k * LSEQ + l) * 16 + (c - 8)]  = val;
      if (c >= 24 && c < 40) Csbuf[((size_t)k * LSEQ + l) * 16 + (c - 24)] = val;
    }
  }
  __syncthreads();

  // GEMM2: A = dts (16x8, K padded to 32 with zeros), B = dtw^T (8x16 per d-tile).
  const v8h zero8 = {};
  v8h dlo = zero8;
  if (hi == 0) dlo = *(const v8h*)&sDts[w][row][0];   // K=0..7 real, rest zero
  const v16h a2 = cat8(dlo, zero8);

  #pragma unroll
  for (int dt = 0; dt < 8; ++dt) {
    v8h wlo = zero8;
    if (hi == 0) wlo = *(const v8h*)&sDtwT[dt * 16 + col16][0];
    v16h b2 = cat8(wlo, zero8);
    v8f acc2 = {};
    acc2 = __builtin_amdgcn_wmma_f32_16x16x32_f16(false, a2, false, b2,
                                                  (short)0, acc2, false, false);
    int dcol = dt * 16 + col16;
    float bia = sBias[dcol];
    #pragma unroll
    for (int r = 0; r < 8; ++r) {
      int l = l0 + r + hi * 8;
      float t = acc2[r] + bia;
      float sp = (t > 20.0f) ? t : log1pf(__expf(t));    // softplus
      deltabuf[((size_t)k * LSEQ + l) * 128 + dcol] = sp;
    }
  }
}

// ---------------------------------------------------------------------------
// Scan phase A: per (k, chunk) block (128 threads = one d per lane).
// Computes the chunk's affine map per state n: h_end = P[n]*h_start + q[n].
// ---------------------------------------------------------------------------
__global__ __launch_bounds__(128)
void scanA_kernel(const float* __restrict__ x,
                  const float* __restrict__ A_logs,   // (12,128,16)
                  const float* __restrict__ deltabuf,
                  const float* __restrict__ Bsbuf,
                  float* __restrict__ Pbuf,           // (12,16,128,16)
                  float* __restrict__ Qbuf)
{
  __shared__ __align__(16) float sB[CHUNK][16];
  const int k = blockIdx.y, chunk = blockIdx.x, d = threadIdx.x;
  const int lbase = chunk * CHUNK;

  const float4* bsrc = (const float4*)(Bsbuf + ((size_t)k * LSEQ + lbase) * 16);
  for (int t = threadIdx.x; t < CHUNK * 16 / 4; t += 128)
    ((float4*)&sB[0][0])[t] = bsrc[t];
  __syncthreads();

  float a_[16];
  #pragma unroll
  for (int n = 0; n < 16; ++n)
    a_[n] = -__expf(A_logs[((size_t)k * 128 + d) * 16 + n]);

  float h[16], P[16];
  #pragma unroll
  for (int n = 0; n < 16; ++n) { h[n] = 0.0f; P[n] = 1.0f; }

  const float* drow = deltabuf + ((size_t)k * LSEQ + lbase) * 128 + d;
  for (int t = 0; t < CHUNK; ++t) {
    int pos = pos_from_l(k, lbase + t);
    float u = x[(size_t)pos * 128 + d];
    float delta = drow[(size_t)t * 128];
    if (t + 8 < CHUNK) __builtin_prefetch(drow + (size_t)(t + 8) * 128, 0, 0);
    float du = delta * u;
    #pragma unroll
    for (int n = 0; n < 16; ++n) {
      float dA = __expf(delta * a_[n]);
      h[n] = dA * h[n] + du * sB[t][n];
      P[n] *= dA;
    }
  }
  size_t obase = (((size_t)k * NCHUNK + chunk) * 128 + d) * 16;
  #pragma unroll
  for (int n = 0; n < 16; ++n) { Pbuf[obase + n] = P[n]; Qbuf[obase + n] = h[n]; }
}

// ---------------------------------------------------------------------------
// Scan phase B: serial combine of 16 chunk maps per (k,d,n) channel.
// ---------------------------------------------------------------------------
__global__ __launch_bounds__(256)
void scanB_kernel(const float* __restrict__ Pbuf,
                  const float* __restrict__ Qbuf,
                  float* __restrict__ Hs)              // chunk-start states
{
  int gid = blockIdx.x * 256 + threadIdx.x;            // (k,d,n) flattened
  int k = gid >> 11;
  int rem = gid & 2047;                                // d*16+n
  float h = 0.0f;
  for (int c = 0; c < NCHUNK; ++c) {
    size_t idx = ((size_t)k * NCHUNK + c) * 2048 + rem;
    Hs[idx] = h;
    h = Pbuf[idx] * h + Qbuf[idx];
  }
}

// ---------------------------------------------------------------------------
// Scan phase C: replay each chunk from its true start state, emit y, and
// scatter y + Ds*u directly into restored (spatial) order.
// ---------------------------------------------------------------------------
__global__ __launch_bounds__(128)
void scanC_kernel(const float* __restrict__ x,
                  const float* __restrict__ A_logs,
                  const float* __restrict__ Ds,        // (12,128)
                  const float* __restrict__ deltabuf,
                  const float* __restrict__ Bsbuf,
                  const float* __restrict__ Csbuf,
                  const float* __restrict__ Hs,
                  float* __restrict__ outscat)         // (12,4096,128), spatial order
{
  __shared__ __align__(16) float sB[CHUNK][16];
  __shared__ __align__(16) float sC[CHUNK][16];
  const int k = blockIdx.y, chunk = blockIdx.x, d = threadIdx.x;
  const int lbase = chunk * CHUNK;

  size_t bcbase = ((size_t)k * LSEQ + lbase) * 16;
  const float4* bsrc = (const float4*)(Bsbuf + bcbase);
  const float4* csrc = (const float4*)(Csbuf + bcbase);
  for (int t = threadIdx.x; t < CHUNK * 16 / 4; t += 128) {
    ((float4*)&sB[0][0])[t] = bsrc[t];
    ((float4*)&sC[0][0])[t] = csrc[t];
  }
  __syncthreads();

  float a_[16], h[16];
  #pragma unroll
  for (int n = 0; n < 16; ++n)
    a_[n] = -__expf(A_logs[((size_t)k * 128 + d) * 16 + n]);
  size_t hbase = (((size_t)k * NCHUNK + chunk) * 128 + d) * 16;
  #pragma unroll
  for (int n = 0; n < 16; ++n) h[n] = Hs[hbase + n];

  const float Dval = Ds[(size_t)k * 128 + d];
  const float* drow = deltabuf + ((size_t)k * LSEQ + lbase) * 128 + d;
  for (int t = 0; t < CHUNK; ++t) {
    int pos = pos_from_l(k, lbase + t);
    float u = x[(size_t)pos * 128 + d];
    float delta = drow[(size_t)t * 128];
    if (t + 8 < CHUNK) __builtin_prefetch(drow + (size_t)(t + 8) * 128, 0, 0);
    float du = delta * u;
    float y = 0.0f;
    #pragma unroll
    for (int n = 0; n < 16; ++n) {
      float dA = __expf(delta * a_[n]);
      h[n] = dA * h[n] + du * sB[t][n];
      y = fmaf(h[n], sC[t][n], y);
    }
    outscat[((size_t)k * LSEQ + pos) * 128 + d] = fmaf(Dval, u, y);
  }
}

// ---------------------------------------------------------------------------
// Final merge over directions (already in spatial order -> pure streaming).
// ---------------------------------------------------------------------------
__global__ __launch_bounds__(256)
void merge_kernel(const float* __restrict__ outscat,
                  const float* __restrict__ merge_w,
                  const float* __restrict__ merge_b,
                  float* __restrict__ out)
{
  int i = blockIdx.x * 256 + threadIdx.x;              // 524288 elements
  float acc = merge_b[0];
  #pragma unroll
  for (int k = 0; k < KDIR; ++k)
    acc = fmaf(merge_w[k], outscat[(size_t)k * LSEQ * 128 + i], acc);
  out[i] = acc;
}

extern "C" void kernel_launch(void* const* d_in, const int* in_sizes, int n_in,
                              void* d_out, int out_size, void* d_ws, size_t ws_size,
                              hipStream_t stream) {
  const float* x      = (const float*)d_in[0];
  const float* xproj  = (const float*)d_in[1];
  const float* dtw    = (const float*)d_in[2];
  const float* dtb    = (const float*)d_in[3];
  const float* A_logs = (const float*)d_in[4];
  const float* Dsp    = (const float*)d_in[5];
  const float* mw     = (const float*)d_in[6];
  const float* mb     = (const float*)d_in[7];
  float* out = (float*)d_out;

  float* wsf = (float*)d_ws;
  size_t off = 0;
  float* deltabuf = wsf + off; off += (size_t)KDIR * LSEQ * DCH;        // 25.2 MB
  float* Bsbuf    = wsf + off; off += (size_t)KDIR * LSEQ * NST;        //  3.1 MB
  float* Csbuf    = wsf + off; off += (size_t)KDIR * LSEQ * NST;        //  3.1 MB
  float* Pbuf     = wsf + off; off += (size_t)KDIR * NCHUNK * DCH * NST;// 1.6 MB
  float* Qbuf     = wsf + off; off += (size_t)KDIR * NCHUNK * DCH * NST;// 1.6 MB
  float* Hsbuf    = wsf + off; off += (size_t)KDIR * NCHUNK * DCH * NST;// 1.6 MB
  float* outscat  = wsf + off; off += (size_t)KDIR * LSEQ * DCH;        // 25.2 MB

  proj_kernel<<<KDIR * 32, 256, 0, stream>>>(x, xproj, dtw, dtb,
                                             deltabuf, Bsbuf, Csbuf);
  scanA_kernel<<<dim3(NCHUNK, KDIR), 128, 0, stream>>>(x, A_logs, deltabuf,
                                                       Bsbuf, Pbuf, Qbuf);
  scanB_kernel<<<(KDIR * DCH * NST) / 256, 256, 0, stream>>>(Pbuf, Qbuf, Hsbuf);
  scanC_kernel<<<dim3(NCHUNK, KDIR), 128, 0, stream>>>(x, A_logs, Dsp, deltabuf,
                                                       Bsbuf, Csbuf, Hsbuf, outscat);
  merge_kernel<<<(LSEQ * DCH) / 256, 256, 0, stream>>>(outscat, mw, mb, out);
}
